// GQA_20306605375750
// MI455X (gfx1250) — matrix operand order repaired and checked by reading, not compiled, not run on hardware
//
#include <hip/hip_runtime.h>
#include <hip/hip_bf16.h>
#include <cstdint>

// Problem constants (match reference)
#define B_  2
#define S_  2048
#define D_  2048
#define H_  16
#define HD_ 128

typedef __attribute__((ext_vector_type(16))) __bf16 v16bf;
typedef __attribute__((ext_vector_type(8)))  float  v8f;
typedef unsigned short u16;

union FragU { v16bf v; uint4 q[2]; };

// ---- gfx1250 async global->LDS path (guarded; falls back to uint4 copy) ----
#if defined(__has_builtin)
#  if __has_builtin(__builtin_amdgcn_global_load_async_to_lds_b128)
#    define USE_ASYNC_LDS 1
#  endif
#  if __has_builtin(__builtin_amdgcn_s_wait_asynccnt)
#    define HAVE_WAIT_ASYNC_BUILTIN 1
#  endif
#endif

typedef int v4i __attribute__((__vector_size__(16)));
#if defined(USE_ASYNC_LDS)
typedef __attribute__((address_space(1))) v4i gv4i;   // global int4
typedef __attribute__((address_space(3))) v4i lv4i;   // LDS int4
#endif

__device__ __forceinline__ void cp_b128(__bf16* dst_lds, const __bf16* src_glob) {
#if defined(USE_ASYNC_LDS)
  __builtin_amdgcn_global_load_async_to_lds_b128((gv4i*)src_glob,
                                                 (lv4i*)dst_lds, 0, 0);
#else
  *(uint4*)dst_lds = *(const uint4*)src_glob;
#endif
}

__device__ __forceinline__ void wait_async0() {
#if defined(HAVE_WAIT_ASYNC_BUILTIN)
  __builtin_amdgcn_s_wait_asynccnt(0);
#elif defined(USE_ASYNC_LDS)
  asm volatile("s_wait_asynccnt 0x0" ::: "memory");
#endif
}

__device__ __forceinline__ v8f v8f_zero() {
  v8f z;
#pragma unroll
  for (int i = 0; i < 8; i++) z[i] = 0.0f;
  return z;
}

__device__ __forceinline__ v8f wmma_bf16(v16bf a, v16bf b, v8f c) {
  return __builtin_amdgcn_wmma_f32_16x16x32_bf16(false, a, false, b, (short)0, c,
                                                 false, false);
}

__device__ __forceinline__ unsigned pk_bf16(float lo, float hi) {
  const unsigned short a = __builtin_bit_cast(unsigned short, (__bf16)lo);
  const unsigned short b = __builtin_bit_cast(unsigned short, (__bf16)hi);
  return ((unsigned)b << 16) | (unsigned)a;
}

// A-fragment: 16 rows (m) x 32-k window from row-major tile (CDNA5 16-bit A layout)
__device__ __forceinline__ v16bf frag_a(const __bf16* base, int stride, int lane) {
  const int m  = lane & 15;
  const int kb = (lane >> 4) << 3;
  FragU f;
  const __bf16* p = base + m * stride + kb;
  f.q[0] = *(const uint4*)(p);
  f.q[1] = *(const uint4*)(p + 16);
  return f.v;
}

// B-fragment: B(k,n) = base[n*stride + k]
__device__ __forceinline__ v16bf frag_b(const __bf16* base, int stride, int lane) {
  const int n  = lane & 15;
  const int kb = (lane >> 4) << 3;
  FragU f;
  const __bf16* p = base + n * stride + kb;
  f.q[0] = *(const uint4*)(p);
  f.q[1] = *(const uint4*)(p + 16);
  return f.v;
}

// ---------------------------------------------------------------------------
// GEMM: C[M,N] f32 = A[M,K] * B[N,K]^T. Tiles 256x128x32; 8 waves as 4(M)x2(N),
// each wave a 64x64 tile (4x4 WMMA accumulators -> 16 ds_load_b128 : 16 wmma).
// ---------------------------------------------------------------------------
template <bool A_F32>
__global__ __launch_bounds__(256) void gemm_bf16_wmma(const void* __restrict__ Ap,
                                                      const float* __restrict__ Bw,
                                                      float* __restrict__ C,
                                                      int M, int N, int K) {
  constexpr int LDT = 40;  // 32 + 8 pad
  __shared__ __bf16 As[256 * LDT];
  __shared__ __bf16 Bs[128 * LDT];

  const int tid  = threadIdx.x;
  const int lane = tid & 31;
  const int w    = tid >> 5;
  const int wm   = w >> 1;  // 0..3 -> rows wm*64
  const int wn   = w & 1;   // 0..1 -> cols wn*64
  const int m0   = blockIdx.y * 256;
  const int n0   = blockIdx.x * 128;

  v8f acc[4][4];
#pragma unroll
  for (int i = 0; i < 4; i++)
#pragma unroll
    for (int j = 0; j < 4; j++) acc[i][j] = v8f_zero();

  for (int k0 = 0; k0 < K; k0 += 32) {
    if constexpr (A_F32) {
      const float* A = (const float*)Ap;
#pragma unroll
      for (int p = 0; p < 8; p++) {
        const int e = p * 1024 + tid * 4;
        const int r = e >> 5, c = e & 31;
        const float* gp = A + (size_t)(m0 + r) * K + k0 + c;
        const float4 f = *(const float4*)gp;
        if (p == 0 && k0 + 32 < K) __builtin_prefetch((const void*)(gp + 32), 0, 0);
        uint2 pk;
        pk.x = pk_bf16(f.x, f.y);
        pk.y = pk_bf16(f.z, f.w);
        *(uint2*)(As + r * LDT + c) = pk;
      }
    } else {
      const __bf16* A = (const __bf16*)Ap;
#pragma unroll
      for (int p = 0; p < 4; p++) {
        const int e = p * 2048 + tid * 8;
        const int r = e >> 5, c = e & 31;
        cp_b128(As + r * LDT + c, A + (size_t)(m0 + r) * K + k0 + c);
      }
    }
#pragma unroll
    for (int p = 0; p < 4; p++) {
      const int e = p * 1024 + tid * 4;
      const int r = e >> 5, c = e & 31;
      const float* gp = Bw + (size_t)(n0 + r) * K + k0 + c;
      const float4 f = *(const float4*)gp;
      if (p == 0 && k0 + 32 < K) __builtin_prefetch((const void*)(gp + 32), 0, 0);
      uint2 pk;
      pk.x = pk_bf16(f.x, f.y);
      pk.y = pk_bf16(f.z, f.w);
      *(uint2*)(Bs + r * LDT + c) = pk;
    }
    if constexpr (!A_F32) wait_async0();
    __syncthreads();

    v16bf a[4];
#pragma unroll
    for (int i = 0; i < 4; i++) a[i] = frag_a(As + (wm * 64 + i * 16) * LDT, LDT, lane);
#pragma unroll
    for (int j = 0; j < 4; j++) {
      const v16bf b = frag_b(Bs + (wn * 64 + j * 16) * LDT, LDT, lane);
#pragma unroll
      for (int i = 0; i < 4; i++) acc[i][j] = wmma_bf16(a[i], b, acc[i][j]);
    }
    __syncthreads();
  }

  const int mr = (lane >> 4) << 3;
  const int nl = lane & 15;
#pragma unroll
  for (int i = 0; i < 4; i++)
#pragma unroll
    for (int j = 0; j < 4; j++)
#pragma unroll
      for (int e = 0; e < 8; e++) {
        const int row = m0 + wm * 64 + i * 16 + mr + e;
        const int col = n0 + wn * 64 + j * 16 + nl;
        C[(size_t)row * N + col] = acc[i][j][e];
      }
}

// ---------------------------------------------------------------------------
// LayerNorm (q,k over full D) + RoPE. Outputs: q,k bf16 [B,H,S,HD];
// V written once as transposed bf16 [B,H,HD,S] so attention staging is pure copy.
// ---------------------------------------------------------------------------
__global__ __launch_bounds__(256) void ln_rope_rk(const float* __restrict__ qkv,
                                                  const float* __restrict__ gq,
                                                  const float* __restrict__ gk,
                                                  u16* __restrict__ qb_,
                                                  u16* __restrict__ kb_,
                                                  u16* __restrict__ vt_) {
  __bf16* qb = (__bf16*)qb_;
  __bf16* kb = (__bf16*)kb_;
  __bf16* vt = (__bf16*)vt_;

  __shared__ float qn[D_];
  __shared__ float kn[D_];
  __shared__ float red[8][4];

  const int tid = threadIdx.x, lane = tid & 31, w = tid >> 5;
  const int row = blockIdx.x;
  const int b = row / S_, s = row % S_;
  const float* base = qkv + (size_t)row * (3 * D_);

  float sq = 0.f, sqq = 0.f, sk = 0.f, skk = 0.f;
#pragma unroll
  for (int j = 0; j < 8; j++) {
    const int i = tid + j * 256;
    const float q = base[i];
    const float k = base[D_ + i];
    qn[i] = q; kn[i] = k;
    sq += q; sqq += q * q; sk += k; skk += k * k;
  }
#pragma unroll
  for (int off = 16; off > 0; off >>= 1) {
    sq  += __shfl_xor(sq,  off, 32);
    sqq += __shfl_xor(sqq, off, 32);
    sk  += __shfl_xor(sk,  off, 32);
    skk += __shfl_xor(skk, off, 32);
  }
  if (lane == 0) { red[w][0] = sq; red[w][1] = sqq; red[w][2] = sk; red[w][3] = skk; }
  __syncthreads();
  float Sq = 0.f, Sqq = 0.f, Sk = 0.f, Skk = 0.f;
#pragma unroll
  for (int i = 0; i < 8; i++) { Sq += red[i][0]; Sqq += red[i][1]; Sk += red[i][2]; Skk += red[i][3]; }
  const float invD = 1.0f / (float)D_;
  const float mq = Sq * invD, vq = Sqq * invD - mq * mq;
  const float mk = Sk * invD, vk = Skk * invD - mk * mk;
  const float rq = rsqrtf(vq + 1e-5f), rk = rsqrtf(vk + 1e-5f);

#pragma unroll
  for (int j = 0; j < 8; j++) {
    const int i = tid + j * 256;
    qn[i] = (qn[i] - mq) * rq * gq[i];
    kn[i] = (kn[i] - mk) * rk * gk[i];
  }
  __syncthreads();

#pragma unroll
  for (int j = 0; j < 8; j++) {
    const int i = tid + j * 256;
    const int h = i >> 7, d = i & 127, dh = d & 63;
    const float inv = __expf(-0.14391156831212772f * (float)dh);  // theta^(-dh/64)
    const float ang = (float)s * inv;
    float c, sn;
    __sincosf(ang, &sn, &c);
    const float xq = qn[i], xk = kn[i];
    const float pq = (d < 64) ? qn[i + 64] : qn[i - 64];
    const float pk = (d < 64) ? kn[i + 64] : kn[i - 64];
    const float oq = (d < 64) ? (xq * c - pq * sn) : (xq * c + pq * sn);
    const float ok = (d < 64) ? (xk * c - pk * sn) : (xk * c + pk * sn);
    const size_t idx = (((size_t)b * H_ + h) * S_ + s) * HD_ + d;
    qb[idx] = (__bf16)oq;
    kb[idx] = (__bf16)ok;
    vt[(((size_t)b * H_ + h) * HD_ + d) * S_ + s] = (__bf16)base[2 * D_ + i];
  }
}

// ---------------------------------------------------------------------------
// Causal flash attention. Grid: (S/128, H, B); 8 waves, each a 16-row q tile.
// 64-key blocks: 16 QK^T + 16 PV WMMAs per block; online softmax per C-frag row.
// K staged [n][d], V staged from pre-transposed global V^T -> pure b128 copies
// (async-to-LDS when available).
// ---------------------------------------------------------------------------
__global__ __launch_bounds__(256) void attn_fwd(const u16* __restrict__ qb_,
                                                const u16* __restrict__ kb_,
                                                const u16* __restrict__ vt_,
                                                u16* __restrict__ ob_) {
  const __bf16* qb  = (const __bf16*)qb_;
  const __bf16* kv  = (const __bf16*)kb_;
  const __bf16* vtp = (const __bf16*)vt_;
  __bf16* ob = (__bf16*)ob_;

  constexpr int LDK = 136;  // 128 + 8
  constexpr int LDV = 72;   // 64 + 8
  constexpr int LDP = 72;
  __shared__ __bf16 Ks[64 * LDK];     // K tile [n][d]
  __shared__ __bf16 Vt[128 * LDV];    // V^T tile [d][n]
  __shared__ __bf16 Ps[8][16 * LDP];  // per-wave P staging [m][n]

  const int tid = threadIdx.x, lane = tid & 31, w = tid >> 5;
  const int qt = blockIdx.x, h = blockIdx.y, b = blockIdx.z;
  const int q0 = qt * 128;
  const size_t bh  = ((size_t)b * H_ + h) * S_;    // q/k row base
  const size_t bhd = ((size_t)b * H_ + h) * HD_;   // v^T row base

  // Preload the wave's Q tile as 4 A-fragments (d-steps of 32).
  v16bf aq[4];
  {
    const __bf16* qrow = qb + (bh + q0 + w * 16) * HD_;
    const int m = lane & 15, kb8 = (lane >> 4) << 3;
#pragma unroll
    for (int ks = 0; ks < 4; ks++) {
      FragU f;
      const __bf16* p = qrow + m * HD_ + ks * 32 + kb8;
      f.q[0] = *(const uint4*)(p);
      f.q[1] = *(const uint4*)(p + 16);
      aq[ks] = f.v;
    }
  }

  float rowmax[8], rowsum[8];
  v8f oacc[8];
#pragma unroll
  for (int e = 0; e < 8; e++) { rowmax[e] = -__builtin_inff(); rowsum[e] = 0.f; }
#pragma unroll
  for (int j = 0; j < 8; j++) oacc[j] = v8f_zero();

  const float scale = 0.08838834764831845f;  // 1/sqrt(128)
  const int nblocks = (q0 + 128) >> 6;       // 64-key blocks, causal bound
  const int mwave_hi = q0 + w * 16 + 15;

  for (int kbI = 0; kbI < nblocks; kbI++) {
    const int n0 = kbI * 64;
    {  // stage K tile [64][128] : 32 elems (4 x b128) per thread
      const int e = tid * 32, r = e >> 7, c = e & 127;
      const __bf16* src = kv + (bh + n0 + r) * HD_ + c;
      __bf16* dst = Ks + r * LDK + c;
#pragma unroll
      for (int i = 0; i < 4; i++) cp_b128(dst + 8 * i, src + 8 * i);
      if (kbI + 1 < nblocks) __builtin_prefetch((const void*)(src + 64 * HD_), 0, 0);
    }
    {  // stage V^T tile [128][64] : 32 elems (4 x b128) per thread
      const int e = tid * 32, r = e >> 6, c = e & 63;
      const __bf16* src = vtp + (bhd + r) * S_ + n0 + c;
      __bf16* dst = Vt + r * LDV + c;
#pragma unroll
      for (int i = 0; i < 4; i++) cp_b128(dst + 8 * i, src + 8 * i);
      if (kbI + 1 < nblocks) __builtin_prefetch((const void*)(src + 64), 0, 0);
    }
    wait_async0();
    __syncthreads();

    if (n0 <= mwave_hi) {  // skip fully-masked blocks for this wave
      // scores: 16(m) x 64(n)
      v8f sc[4];
#pragma unroll
      for (int jn = 0; jn < 4; jn++) sc[jn] = v8f_zero();
#pragma unroll
      for (int ks = 0; ks < 4; ks++) {
#pragma unroll
        for (int jn = 0; jn < 4; jn++) {
          const v16bf bk = frag_b(Ks + (jn * 16) * LDK + ks * 32, LDK, lane);
          sc[jn] = wmma_bf16(aq[ks], bk, sc[jn]);
        }
      }

      const int mbase = q0 + w * 16 + ((lane >> 4) << 3);
      const int nc = n0 + (lane & 15);
#pragma unroll
      for (int e = 0; e < 8; e++) {
        float x[4];
#pragma unroll
        for (int jn = 0; jn < 4; jn++) {
          float v = sc[jn][e] * scale;
          if (nc + jn * 16 > mbase + e) v = -__builtin_inff();
          x[jn] = v;
        }
        float lm = fmaxf(fmaxf(x[0], x[1]), fmaxf(x[2], x[3]));
#pragma unroll
        for (int off = 1; off < 16; off <<= 1) lm = fmaxf(lm, __shfl_xor(lm, off, 32));
        const float mnew = fmaxf(rowmax[e], lm);
        const float r = __expf(rowmax[e] - mnew);
        float p[4], ps = 0.f;
#pragma unroll
        for (int jn = 0; jn < 4; jn++) { p[jn] = __expf(x[jn] - mnew); ps += p[jn]; }
#pragma unroll
        for (int off = 1; off < 16; off <<= 1) ps += __shfl_xor(ps, off, 32);
        rowsum[e] = rowsum[e] * r + ps;
        rowmax[e] = mnew;
#pragma unroll
        for (int j = 0; j < 8; j++) oacc[j][e] *= r;
        __bf16* pr = Ps[w] + (((lane >> 4) << 3) + e) * LDP + (lane & 15);
#pragma unroll
        for (int jn = 0; jn < 4; jn++) pr[jn * 16] = (__bf16)p[jn];
      }

      // PV: O(16x128) += P(16x64) * V(64x128)
      const v16bf ap0 = frag_a(Ps[w], LDP, lane);
      const v16bf ap1 = frag_a(Ps[w] + 32, LDP, lane);
#pragma unroll
      for (int j = 0; j < 8; j++) {
        const v16bf bv0 = frag_b(Vt + (j * 16) * LDV, LDV, lane);
        const v16bf bv1 = frag_b(Vt + (j * 16) * LDV + 32, LDV, lane);
        oacc[j] = wmma_bf16(ap0, bv0, oacc[j]);
        oacc[j] = wmma_bf16(ap1, bv1, oacc[j]);
      }
    }
    __syncthreads();
  }

  const int mr = (lane >> 4) << 3, nl = lane & 15;
#pragma unroll
  for (int e = 0; e < 8; e++) {
    const float inv = 1.0f / rowsum[e];
    const int row = q0 + w * 16 + mr + e;
#pragma unroll
    for (int j = 0; j < 8; j++) {
      const int col = h * HD_ + j * 16 + nl;
      ob[((size_t)b * S_ + row) * D_ + col] = (__bf16)(oacc[j][e] * inv);
    }
  }
}

// ---------------------------------------------------------------------------
extern "C" void kernel_launch(void* const* d_in, const int* in_sizes, int n_in,
                              void* d_out, int out_size, void* d_ws, size_t ws_size,
                              hipStream_t stream) {
  (void)in_sizes; (void)n_in; (void)out_size; (void)ws_size;
  const float* x    = (const float*)d_in[0];
  const float* Wqkv = (const float*)d_in[1];
  const float* gq   = (const float*)d_in[2];
  const float* gk   = (const float*)d_in[3];
  const float* Wout = (const float*)d_in[4];
  float* out = (float*)d_out;

  const size_t rows = (size_t)B_ * S_;  // 4096
  char* ws = (char*)d_ws;
  float* qkv = (float*)ws;  ws += rows * 3 * D_ * sizeof(float);  // 100.7 MB
  u16* qb    = (u16*)ws;    ws += rows * D_ * sizeof(u16);        // 16.8 MB
  u16* kb    = (u16*)ws;    ws += rows * D_ * sizeof(u16);
  u16* vt    = (u16*)ws;    ws += rows * D_ * sizeof(u16);
  u16* attn  = (u16*)ws;    ws += rows * D_ * sizeof(u16);

  const dim3 blk(256);
  // 1) qkv = x @ Wqkv^T   (M=4096, N=6144, K=2048)
  gemm_bf16_wmma<true><<<dim3(3 * D_ / 128, (unsigned)(rows / 256)), blk, 0, stream>>>(
      (const void*)x, Wqkv, qkv, (int)rows, 3 * D_, D_);
  // 2) LN + RoPE + bf16 pack (V pre-transposed)
  ln_rope_rk<<<dim3((unsigned)rows), blk, 0, stream>>>(qkv, gq, gk, qb, kb, vt);
  // 3) causal flash attention
  attn_fwd<<<dim3(S_ / 128, H_, B_), blk, 0, stream>>>(qb, kb, vt, attn);
  // 4) out = attn @ Wout^T (M=4096, N=2048, K=2048)
  gemm_bf16_wmma<false><<<dim3(D_ / 128, (unsigned)(rows / 256)), blk, 0, stream>>>(
      (const void*)attn, Wout, out, (int)rows, D_, D_);
}
